// RelLearnableMultiHeadAttn_10771777978441
// MI455X (gfx1250) — matrix operand order repaired and checked by reading, not compiled, not run on hardware
//
#include <hip/hip_runtime.h>

// Transformer-XL relative attention score kernel for MI455X (gfx1250, wave32).
// AC = (Q + r_w_bias) K^T ; B_ = Q R^T ; out = AC + rel_shift(B_ + r_bias)
// Strategy: bf16 hi/lo split (3-term) on v_wmma_f32_16x16x32_bf16, f32 accum.
//
// v4: 64-row output stripe per WG (fixed (b,n)). One shared rwq A-fragment set
//     (4 row-blocks, 128 VGPRs -> stays under the 256-VGPR MSB boundary)
//     drives BOTH GEMMs: (Q+rwb)R^T = B_ + rwb.R^T, corrected via
//     radj[j] = r_bias[j,n] - dot(rwb[n], r_emb[j,n]). The single extra X row
//     (i0+64, needed by the rel_shift) is computed with scalar f32 dots in the
//     pre-phase instead of a 15/16-wasted WMMA row-block. Branchless shifted
//     gather in the epilogue. X stripe (65x1024 f32 = 266KB) + radj (4KB) fit
//     CDNA5's 320KB/WGP LDS.

#define QLEN 1024
#define KLEN 1024
#define BSZ  4
#define NH   16
#define DH   64

typedef __bf16 bf16_t;
typedef bf16_t v16bf __attribute__((ext_vector_type(16)));
typedef float  v8f   __attribute__((ext_vector_type(8)));

// A-matrix (16x32 bf16, MxK) fragment: lane half 0 holds K {0..7,16..23},
// half 1 holds K {8..15,24..31}; VGPR v packs pairs (per ISA 7.12.2).
static __device__ __forceinline__ void make_frag_a(const float* __restrict__ row,
                                                   const float* __restrict__ bias,
                                                   int kb, int half,
                                                   v16bf& hi, v16bf& lo) {
#pragma unroll
    for (int v = 0; v < 8; ++v) {
        int d = 32 * kb + ((v & 4) ? 16 : 0) + half * 8 + (v & 3) * 2;
        float a0 = row[d] + bias[d];
        float a1 = row[d + 1] + bias[d + 1];
        bf16_t h0 = (bf16_t)a0;
        bf16_t h1 = (bf16_t)a1;
        hi[2 * v]     = h0;
        hi[2 * v + 1] = h1;
        lo[2 * v]     = (bf16_t)(a0 - (float)h0);
        lo[2 * v + 1] = (bf16_t)(a1 - (float)h1);
    }
}

// B-matrix (32x16 bf16, KxN) fragment: lane half 0 holds K {0..7,16..23},
// half 1 holds K {8..15,24..31}, matching the A-fragment k order.
static __device__ __forceinline__ void make_frag_b(const float* __restrict__ row,
                                                   int kb, int half,
                                                   v16bf& hi, v16bf& lo) {
#pragma unroll
    for (int e = 0; e < 16; ++e) {
        int d = 32 * kb + ((e & 8) ? 16 : 0) + half * 8 + (e & 7);
        float a = row[d];
        bf16_t h = (bf16_t)a;
        hi[e] = h;
        lo[e] = (bf16_t)(a - (float)h);
    }
}

// Three-term split product accumulate: acc += A*B with A=ah+al, B=bh+bl
// (dropping lo*lo; ~2^-17 relative error, close to f32 reference).
static __device__ __forceinline__ v8f wmma3(v8f acc, v16bf ah, v16bf al,
                                            v16bf bh, v16bf bl) {
    acc = __builtin_amdgcn_wmma_f32_16x16x32_bf16(false, ah, false, bh, (short)0, acc, false, false);
    acc = __builtin_amdgcn_wmma_f32_16x16x32_bf16(false, ah, false, bl, (short)0, acc, false, false);
    acc = __builtin_amdgcn_wmma_f32_16x16x32_bf16(false, al, false, bh, (short)0, acc, false, false);
    return acc;
}

__global__ __launch_bounds__(256) void relattn_kernel(
    const float* __restrict__ q,      // (QLEN, BSZ, NH, DH)
    const float* __restrict__ k,      // (KLEN, BSZ, NH, DH)
    const float* __restrict__ remb,   // (KLEN, NH, DH)
    const float* __restrict__ rwb,    // (NH, DH)
    const float* __restrict__ rbias,  // (KLEN, NH)
    float* __restrict__ out)          // (QLEN, KLEN, BSZ, NH)
{
    const int tid  = threadIdx.x;
    const int wave = tid >> 5;
    const int lane = tid & 31;
    const int half = lane >> 4;
    const int lm   = lane & 15;
    const int i0   = blockIdx.x * 64;     // output row stripe [i0, i0+64)
    const int bn   = blockIdx.y;
    const int b    = bn >> 4;
    const int n    = bn & 15;

    // X stripe: rows i0..i0+64 inclusive (65 rows) of B_ + r_bias, f32.
    __shared__ float Xs[65 * 1024];       // 266,240 B
    __shared__ float radj_s[KLEN];        // 4,096 B  (total < 320KB WGP LDS)

    const float* biasrow = rwb + n * DH;

    // ---- Pre-phase (scalar f32):
    //  radj[j]   = r_bias[j,n] - dot(rwb[n,:], r_emb[j,n,:])
    //  Xs[64][j] = dot(q[i0+64], r_emb[j,n,:]) + r_bias[j,n]   (rel_shift row)
    {
        int g64 = i0 + 64;
        if (g64 > QLEN - 1) g64 = QLEN - 1;    // clamp: row 1024 never read
        const float* q64 = q + (((size_t)g64 * BSZ + b) * NH + n) * DH;
#pragma unroll
        for (int j4 = 0; j4 < 4; ++j4) {
            int jj = tid + 256 * j4;
            const float* rrow = remb + ((size_t)jj * NH + n) * DH;
            float sadj = 0.f, s64 = 0.f;
#pragma unroll
            for (int d = 0; d < DH; ++d) {
                float rv = rrow[d];
                sadj = fmaf(biasrow[d], rv, sadj);
                s64  = fmaf(q64[d],     rv, s64);
            }
            float rb_j = rbias[(size_t)jj * NH + n];
            radj_s[jj]          = rb_j - sadj;
            Xs[64 * 1024 + jj]  = s64 + rb_j;
        }
    }

    // ---- Preload shared rwq A-fragments for 4 row-blocks (128 VGPRs) ----
    v16bf ah[4][2], al[4][2];
#pragma unroll
    for (int rb = 0; rb < 4; ++rb) {
        int grow = i0 + 16 * rb + lm;          // always < QLEN
        const float* qrow = q + (((size_t)grow * BSZ + b) * NH + n) * DH;
        make_frag_a(qrow, biasrow, 0, half, ah[rb][0], al[rb][0]);
        make_frag_a(qrow, biasrow, 1, half, ah[rb][1], al[rb][1]);
    }
    __syncthreads();   // radj_s / Xs row 64 ready

    // ---------------- Phase A: X = rwq.R^T + radj stripe -> LDS ----------------
    for (int j = 0; j < 8; ++j) {
        int ct   = wave * 8 + j;
        int jrow = 16 * ct + lm;               // B column owned by this lane
        const float* rrow = remb + ((size_t)jrow * NH + n) * DH;
        v16bf bh0, bl0, bh1, bl1;              // converted ONCE, reused 4x
        make_frag_b(rrow, 0, half, bh0, bl0);
        make_frag_b(rrow, 1, half, bh1, bl1);
        float radd = radj_s[jrow];

#pragma unroll
        for (int rb2 = 0; rb2 < 4; ++rb2) {
            v8f acc = {0.f, 0.f, 0.f, 0.f, 0.f, 0.f, 0.f, 0.f};
            acc = wmma3(acc, ah[rb2][0], al[rb2][0], bh0, bl0);
            acc = wmma3(acc, ah[rb2][1], al[rb2][1], bh1, bl1);
#pragma unroll
            for (int v = 0; v < 8; ++v) {
                int row_l = 16 * rb2 + v + 8 * half;
                Xs[row_l * 1024 + jrow] = acc[v] + radd;   // unconditional
            }
        }
    }
    __syncthreads();

    // ---------------- Phase B: AC + rel_shift gather (same A frags) ----------------
    for (int j = 0; j < 8; ++j) {
        int ct   = wave * 8 + j;
        int jrow = 16 * ct + lm;
        const float* krow = k + (((size_t)jrow * BSZ + b) * NH + n) * DH;
        v16bf bh0, bl0, bh1, bl1;              // converted ONCE, reused 4x
        make_frag_b(krow, 0, half, bh0, bl0);
        make_frag_b(krow, 1, half, bh1, bl1);
        const int r = jrow;

#pragma unroll
        for (int rb = 0; rb < 4; ++rb) {
            v8f acc = {0.f, 0.f, 0.f, 0.f, 0.f, 0.f, 0.f, 0.f};
            acc = wmma3(acc, ah[rb][0], al[rb][0], bh0, bl0);
            acc = wmma3(acc, ah[rb][1], al[rb][1], bh1, bl1);
#pragma unroll
            for (int v = 0; v < 8; ++v) {
                int p_l   = 16 * rb + v + 8 * half;
                int p     = i0 + p_l;
                bool low  = (r <= p);
                int  srow = low ? p_l : (p_l + 1);                 // x[p,..] / x[p+1,..]
                int  scol = low ? (KLEN - 1 - p + r) : (r - p - 2);// scol=-1 (r==p+1) stays in-bounds
                float bd  = Xs[srow * 1024 + scol];
                bd = (r == p + 1) ? 0.0f : bd;                     // shifted-in zero pad
                out[((size_t)p * KLEN + r) * (BSZ * NH) + b * NH + n] = acc[v] + bd;
            }
        }
    }
}

extern "C" void kernel_launch(void* const* d_in, const int* in_sizes, int n_in,
                              void* d_out, int out_size, void* d_ws, size_t ws_size,
                              hipStream_t stream) {
    const float* q     = (const float*)d_in[0];
    const float* k     = (const float*)d_in[1];
    const float* remb  = (const float*)d_in[2];
    const float* rwb   = (const float*)d_in[3];
    const float* rbias = (const float*)d_in[4];
    float* out = (float*)d_out;

    dim3 grid(QLEN / 64, BSZ * NH);   // 16 row-stripes x 64 (b,n) pairs
    relattn_kernel<<<grid, 256, 0, stream>>>(q, k, remb, rwb, rbias, out);
}
